// ConvKAN_30159260352768
// MI455X (gfx1250) — compile-verified, matrix operand order checked by reading, not run on hardware
//
#include <hip/hip_runtime.h>
#include <hip/hip_bf16.h>
#include <math.h>

typedef __attribute__((ext_vector_type(16))) _Float16 v16h;
typedef __attribute__((ext_vector_type(8)))  float    v8f;

// ---------------------------------------------------------------------------
// Problem constants
// ---------------------------------------------------------------------------
#define BATCH   256
// conv1
#define C1_G    5
#define C1_OC   13
#define C1_T    10          // 2 trig * 1 in-ch * 5 g
// conv2
#define C2_G    8
#define C2_IC   13
#define C2_OC   21
#define C2_T    208         // 2 * 13 * 8
#define C2_TP   224         // padded to multiple of 32
#define C2_KW   13
#define C2_POS  169         // 13*13
#define C2_M    32          // padded output channels
#define NPIX    196         // 14*14
#define NPIX_ALL (BATCH * NPIX)   // 50176
// fk1
#define FK1_D   1029
#define FK1_G   13
#define FK1_K   26754       // 1029*13*2
#define FK1_KP  26784       // padded to multiple of 32 (837*32)
#define FK1_O   89
#define FK1_OP  96
// fk2
#define FK2_G   8
#define FK2_O   10

// ---------------------------------------------------------------------------
// Stage 1a: fold conv1 weights  [2,13,1,5,5,5] -> w1f[oc][kh][kw][t], t=trig*5+g
// ---------------------------------------------------------------------------
__global__ void k_fold_w1(const float* __restrict__ c1, float* __restrict__ w1f) {
    int i = blockIdx.x * blockDim.x + threadIdx.x;
    if (i >= C1_OC * 5 * 5 * C1_T) return;
    int t   = i % C1_T;
    int p   = i / C1_T;
    int kw  = p % 5;  p /= 5;
    int kh  = p % 5;
    int oc  = p / 5;
    int trig = t / C1_G, g = t % C1_G;
    // c1 strides: trig 1625, oc 125, kh 25, kw 5, g 1
    w1f[i] = c1[trig * 1625 + oc * 125 + kh * 25 + kw * 5 + g];
}

// ---------------------------------------------------------------------------
// Stage 1b: trig expand input -> feat1[b][y][x][t]  (f32, channels-last)
// ---------------------------------------------------------------------------
__global__ void k_trig1(const float* __restrict__ x, float* __restrict__ feat1) {
    int i = blockIdx.x * blockDim.x + threadIdx.x;
    if (i >= BATCH * 28 * 28 * C1_T) return;
    int t = i % C1_T;
    int p = i / C1_T;                 // b*784 + y*28 + x
    float v = x[p];
    float g = (float)(t % C1_G + 1);
    feat1[i] = (t < C1_G) ? __cosf(g * v) : __sinf(g * v);
}

// ---------------------------------------------------------------------------
// Stage 1c: conv1 (5x5, pad 2) + bias + maxpool2 -> h1[b][ph][pw][c] f32
// ---------------------------------------------------------------------------
__global__ void k_conv1_pool(const float* __restrict__ feat1,
                             const float* __restrict__ w1f,
                             const float* __restrict__ bias1,
                             float* __restrict__ h1) {
    int i = blockIdx.x * blockDim.x + threadIdx.x;
    if (i >= BATCH * 14 * 14 * C1_OC) return;
    int oc = i % C1_OC;
    int p  = i / C1_OC;
    int pw = p % 14; p /= 14;
    int ph = p % 14;
    int b  = p / 14;

    float m = -INFINITY;
    for (int dy = 0; dy < 2; ++dy)
    for (int dx = 0; dx < 2; ++dx) {
        int oy = ph * 2 + dy, ox = pw * 2 + dx;
        float s = 0.f;
        for (int kh = 0; kh < 5; ++kh) {
            int iy = oy + kh - 2;
            if (iy < 0 || iy >= 28) continue;
            for (int kw = 0; kw < 5; ++kw) {
                int ix = ox + kw - 2;
                if (ix < 0 || ix >= 28) continue;
                const float* f = feat1 + ((b * 28 + iy) * 28 + ix) * C1_T;
                const float* w = w1f + ((oc * 5 + kh) * 5 + kw) * C1_T;
                #pragma unroll
                for (int t = 0; t < C1_T; ++t) s = fmaf(f[t], w[t], s);
            }
        }
        m = fmaxf(m, s);
    }
    h1[i] = m + bias1[oc];
}

// ---------------------------------------------------------------------------
// Stage 2a: fold conv2 weights -> w2f[oc(32)][pos(169)][t(224)] f16, zero pad
// ---------------------------------------------------------------------------
__global__ void k_fold_w2(const float* __restrict__ c2, _Float16* __restrict__ w2f) {
    int i = blockIdx.x * blockDim.x + threadIdx.x;
    if (i >= C2_M * C2_POS * C2_TP) return;
    int t   = i % C2_TP;
    int p   = i / C2_TP;
    int pos = p % C2_POS;
    int oc  = p / C2_POS;
    float v = 0.f;
    if (oc < C2_OC && t < C2_T) {
        int trig = t / 104, r = t % 104, c = r / 8, g = r % 8;
        int kh = pos / C2_KW, kw = pos % C2_KW;
        // c2 strides: trig 369096, oc 17576, c 1352, kh 104, kw 8, g 1
        v = c2[trig * 369096 + oc * 17576 + c * 1352 + kh * 104 + kw * 8 + g];
    }
    w2f[i] = (_Float16)v;
}

// ---------------------------------------------------------------------------
// Stage 2b: trig expand h1 -> feat2[b][y][x][t(224)] f16 (channels-last, padded)
// ---------------------------------------------------------------------------
__global__ void k_trig2(const float* __restrict__ h1, _Float16* __restrict__ feat2) {
    int i = blockIdx.x * blockDim.x + threadIdx.x;
    if (i >= BATCH * NPIX * C2_TP) return;
    int t = i % C2_TP;
    int p = i / C2_TP;              // b*196 + y*14 + x
    float v = 0.f;
    if (t < C2_T) {
        int trig = t / 104, r = t % 104, c = r / 8;
        float g = (float)(r % 8 + 1);
        float hv = h1[p * C2_IC + c];
        v = trig ? __sinf(g * hv) : __cosf(g * hv);
    }
    feat2[i] = (_Float16)v;
}

// ---------------------------------------------------------------------------
// Stage 2c: conv2 as shifted WMMA GEMM.
//   M = 32 padded oc, N = 50176 pixels, K = 169 positions x 224 channels.
//   Per workgroup: 256 pixels (8 waves x 32); per wave: 2 pixel columns x
//   2 oc tiles = 4 accumulators -> 4 WMMAs per staged A-fragment pair.
//   A tile (32x224, 14.3 KB) staged in LDS per kernel position; cooperative
//   copy offsets hoisted out of the position loop.
// ---------------------------------------------------------------------------
__global__ __launch_bounds__(256) void k_conv2_wmma(
        const _Float16* __restrict__ feat2,
        const _Float16* __restrict__ w2f,
        float* __restrict__ out2) {
    __shared__ _Float16 aw[C2_M * C2_TP];          // 14336 B

    const int tid   = threadIdx.x;
    const int lane  = tid & 31;
    const int l15   = lane & 15;
    const int wv    = tid >> 5;                    // wave in workgroup, 0..7
    const int khalf = (lane >> 4) * 16;            // K half: lane half h holds contiguous k = 16h + s

    // two output pixels per lane column
    const int p0 = blockIdx.x * 256 + wv * 32 + l15;
    const int p1 = p0 + 16;
    const int b0 = p0 / NPIX, q0 = p0 % NPIX, oy0 = q0 / 14, ox0 = q0 % 14;
    const int b1 = p1 / NPIX, q1 = p1 % NPIX, oy1 = q1 / 14, ox1 = q1 % 14;
    const _Float16* f2b0 = feat2 + (size_t)b0 * (NPIX * C2_TP) + khalf;
    const _Float16* f2b1 = feat2 + (size_t)b1 * (NPIX * C2_TP) + khalf;

    // hoisted cooperative-copy offsets: 896 uint4 chunks, 3.5 per thread
    int srcB[4], dstB[4];
    #pragma unroll
    for (int it = 0; it < 4; ++it) {
        int j    = tid + it * 256;                 // valid when < 896
        int oc   = j / 28;                         // 28 = C2_TP/8
        int toff = (j - oc * 28) * 8;
        srcB[it] = oc * (C2_POS * C2_TP) + toff;
        dstB[it] = oc * C2_TP + toff;
    }

    v8f c00 = {}, c01 = {}, c10 = {}, c11 = {};

    int pos = 0;
    for (int kh = 0; kh < C2_KW; ++kh) {
        const int iy0 = oy0 + kh - 6;
        const int iy1 = oy1 + kh - 6;
        for (int kw = 0; kw < C2_KW; ++kw, ++pos) {
            // stage A tile for this position
            const _Float16* wsrc = w2f + pos * C2_TP;
            #pragma unroll
            for (int it = 0; it < 3; ++it)
                *(uint4*)(&aw[dstB[it]]) = *(const uint4*)(&wsrc[srcB[it]]);
            if (tid < 128)
                *(uint4*)(&aw[dstB[3]]) = *(const uint4*)(&wsrc[srcB[3]]);
            __syncthreads();

            const int ix0 = ox0 + kw - 6;
            const int ix1 = ox1 + kw - 6;
            const bool v0 = (iy0 >= 0) & (iy0 < 14) & (ix0 >= 0) & (ix0 < 14);
            const bool v1 = (iy1 >= 0) & (iy1 < 14) & (ix1 >= 0) & (ix1 < 14);
            const _Float16* bp0 = f2b0 + (iy0 * 14 + ix0) * C2_TP;
            const _Float16* bp1 = f2b1 + (iy1 * 14 + ix1) * C2_TP;

            #pragma unroll
            for (int kc = 0; kc < C2_TP / 32; ++kc) {
                v16h a0 = *(const v16h*)(&aw[l15 * C2_TP + kc * 32 + khalf]);
                v16h a1 = *(const v16h*)(&aw[(l15 + 16) * C2_TP + kc * 32 + khalf]);
                v16h bf0 = {};
                if (v0) bf0 = *(const v16h*)(bp0 + kc * 32);     // halo -> zeros
                v16h bf1 = {};
                if (v1) bf1 = *(const v16h*)(bp1 + kc * 32);
                c00 = __builtin_amdgcn_wmma_f32_16x16x32_f16(false, a0, false, bf0,
                                                             (short)0, c00, false, false);
                c10 = __builtin_amdgcn_wmma_f32_16x16x32_f16(false, a1, false, bf0,
                                                             (short)0, c10, false, false);
                c01 = __builtin_amdgcn_wmma_f32_16x16x32_f16(false, a0, false, bf1,
                                                             (short)0, c01, false, false);
                c11 = __builtin_amdgcn_wmma_f32_16x16x32_f16(false, a1, false, bf1,
                                                             (short)0, c11, false, false);
            }
            __syncthreads();
        }
    }

    // C layout: VGPR r, lanes 0-15 -> M=r, lanes 16-31 -> M=r+8; N = lane%16.
    const int mhi = (lane >> 4) * 8;
    #pragma unroll
    for (int r = 0; r < 8; ++r) {
        int ocA = r + mhi;                 // tile 0: oc 0..15
        if (ocA < C2_OC) {
            out2[((size_t)b0 * C2_OC + ocA) * NPIX + q0] = c00[r];
            out2[((size_t)b1 * C2_OC + ocA) * NPIX + q1] = c01[r];
        }
        int ocB = 16 + r + mhi;            // tile 1: oc 16..31 (valid 16..20)
        if (ocB < C2_OC) {
            out2[((size_t)b0 * C2_OC + ocB) * NPIX + q0] = c10[r];
            out2[((size_t)b1 * C2_OC + ocB) * NPIX + q1] = c11[r];
        }
    }
}

// ---------------------------------------------------------------------------
// Stage 2d: bias + maxpool2 + flatten -> h2[b][1029] (c*49 + ph*7 + pw)
// ---------------------------------------------------------------------------
__global__ void k_pool2_flat(const float* __restrict__ out2,
                             const float* __restrict__ bias2,
                             float* __restrict__ h2) {
    int i = blockIdx.x * blockDim.x + threadIdx.x;
    if (i >= BATCH * C2_OC * 49) return;
    int pw = i % 7; int p = i / 7;
    int ph = p % 7; p /= 7;
    int c  = p % C2_OC;
    int b  = p / C2_OC;
    const float* src = out2 + ((size_t)b * C2_OC + c) * NPIX;
    float m = -INFINITY;
    #pragma unroll
    for (int dy = 0; dy < 2; ++dy)
    #pragma unroll
    for (int dx = 0; dx < 2; ++dx)
        m = fmaxf(m, src[(ph * 2 + dy) * 14 + (pw * 2 + dx)]);
    h2[(size_t)b * FK1_D + c * 49 + ph * 7 + pw] = m + bias2[c];
}

// ---------------------------------------------------------------------------
// Stage 3a: trig features for fk1 -> feat3[b][KP] f16 (k = d*26 + trig*13 + g-1)
// ---------------------------------------------------------------------------
__global__ void k_trig3(const float* __restrict__ h2, _Float16* __restrict__ feat3) {
    int i = blockIdx.x * blockDim.x + threadIdx.x;
    if (i >= BATCH * FK1_KP) return;
    int k = i % FK1_KP;
    int b = i / FK1_KP;
    float v = 0.f;
    if (k < FK1_K) {
        int d = k / 26, r = k % 26;
        float g = (float)(r % FK1_G + 1);
        float hv = h2[(size_t)b * FK1_D + d];
        v = (r < FK1_G) ? __cosf(g * hv) : __sinf(g * hv);
    }
    feat3[i] = (_Float16)v;
}

// ---------------------------------------------------------------------------
// Stage 3b: fold fk1 weights -> wf1[o(96)][KP] f16, zero pad
// ---------------------------------------------------------------------------
__global__ void k_fold_w3(const float* __restrict__ fk1c, _Float16* __restrict__ wf1) {
    int i = blockIdx.x * blockDim.x + threadIdx.x;
    if (i >= FK1_OP * FK1_KP) return;
    int k = i % FK1_KP;
    int o = i / FK1_KP;
    float v = 0.f;
    if (o < FK1_O && k < FK1_K) {
        int d = k / 26, r = k % 26;
        int trig = r / FK1_G, g = r % FK1_G;
        // fk1c strides: trig 1190553 (=89*1029*13), o 13377 (=1029*13), d 13, g 1
        v = fk1c[trig * 1190553 + o * 13377 + d * 13 + g];
    }
    wf1[i] = (_Float16)v;
}

// ---------------------------------------------------------------------------
// Stage 3c: fk1 WMMA GEMM: [96 x KP] x [KP x 256] -> y1[b][89]
// 96 waves total (6 M-tiles x 16 N-tiles), 12 blocks x 8 waves.
// ---------------------------------------------------------------------------
__global__ __launch_bounds__(256) void k_fk1_wmma(
        const _Float16* __restrict__ feat3,
        const _Float16* __restrict__ wf1,
        const float* __restrict__ bias,
        float* __restrict__ y1) {
    const int tid  = threadIdx.x;
    const int lane = tid & 31;
    const int wg   = blockIdx.x * 8 + (tid >> 5);   // 0..95
    const int mt   = wg % 6;
    const int nt   = wg / 6;
    const int khalf = (lane >> 4) * 16;
    const _Float16* ap = wf1   + (size_t)(mt * 16 + (lane & 15)) * FK1_KP + khalf;
    const _Float16* bp = feat3 + (size_t)(nt * 16 + (lane & 15)) * FK1_KP + khalf;

    v8f c = {};
    for (int kc = 0; kc < FK1_KP / 32; ++kc) {      // 837 chunks
        v16h a  = *(const v16h*)(ap + kc * 32);
        v16h bf = *(const v16h*)(bp + kc * 32);
        c = __builtin_amdgcn_wmma_f32_16x16x32_f16(false, a, false, bf,
                                                   (short)0, c, false, false);
    }

    const int bb  = nt * 16 + (lane & 15);
    const int mhi = (lane >> 4) * 8;
    #pragma unroll
    for (int r = 0; r < 8; ++r) {
        int o = mt * 16 + r + mhi;
        if (o < FK1_O)
            y1[(size_t)bb * FK1_O + o] = c[r] + bias[o];
    }
}

// ---------------------------------------------------------------------------
// Stage 4: fk2 (tiny) -> out[b][10] f32
// ---------------------------------------------------------------------------
__global__ void k_fk2(const float* __restrict__ y1,
                      const float* __restrict__ fk2c,
                      const float* __restrict__ bias,
                      float* __restrict__ out) {
    int i = blockIdx.x * blockDim.x + threadIdx.x;
    if (i >= BATCH * FK2_O) return;
    int o = i % FK2_O;
    int b = i / FK2_O;
    float acc = bias[o];
    for (int d = 0; d < FK1_O; ++d) {
        float v = y1[(size_t)b * FK1_O + d];
        #pragma unroll
        for (int g = 1; g <= FK2_G; ++g) {
            float gv = (float)g * v;
            // fk2c strides: trig 7120 (=10*89*8), o 712, d 8, g 1
            acc = fmaf(__cosf(gv), fk2c[o * 712 + d * 8 + (g - 1)], acc);
            acc = fmaf(__sinf(gv), fk2c[7120 + o * 712 + d * 8 + (g - 1)], acc);
        }
    }
    out[(size_t)b * FK2_O + o] = acc;
}

// ---------------------------------------------------------------------------
// Host launcher
// ---------------------------------------------------------------------------
static inline int cdiv(long long n, int d) { return (int)((n + d - 1) / d); }

extern "C" void kernel_launch(void* const* d_in, const int* in_sizes, int n_in,
                              void* d_out, int out_size, void* d_ws, size_t ws_size,
                              hipStream_t stream) {
    const float* x      = (const float*)d_in[0];
    const float* c1c    = (const float*)d_in[1];
    const float* c1b    = (const float*)d_in[2];
    const float* c2c    = (const float*)d_in[3];
    const float* c2b    = (const float*)d_in[4];
    const float* fk1c   = (const float*)d_in[5];
    const float* fk1b   = (const float*)d_in[6];
    const float* fk2c   = (const float*)d_in[7];
    const float* fk2b   = (const float*)d_in[8];
    float* out = (float*)d_out;

    // workspace carve-up (256 B aligned)
    char* ws = (char*)d_ws;
    size_t off = 0;
    auto take = [&](size_t bytes) -> char* {
        char* p = ws + off;
        off = (off + bytes + 255) & ~(size_t)255;
        return p;
    };
    float*    w1f   = (float*)   take((size_t)C1_OC * 25 * C1_T * 4);
    float*    feat1 = (float*)   take((size_t)BATCH * 784 * C1_T * 4);
    float*    h1    = (float*)   take((size_t)BATCH * NPIX * C1_OC * 4);
    _Float16* w2f   = (_Float16*)take((size_t)C2_M * C2_POS * C2_TP * 2);
    _Float16* feat2 = (_Float16*)take((size_t)BATCH * NPIX * C2_TP * 2);
    float*    out2  = (float*)   take((size_t)BATCH * C2_OC * NPIX * 4);
    float*    h2    = (float*)   take((size_t)BATCH * FK1_D * 4);
    _Float16* feat3 = (_Float16*)take((size_t)BATCH * FK1_KP * 2);
    _Float16* wf1   = (_Float16*)take((size_t)FK1_OP * FK1_KP * 2);
    float*    y1    = (float*)   take((size_t)BATCH * FK1_O * 4);
    (void)ws_size; (void)n_in; (void)in_sizes; (void)out_size;

    const int TB = 256;

    // stage 1: conv1
    k_fold_w1<<<cdiv(C1_OC * 25 * C1_T, TB), TB, 0, stream>>>(c1c, w1f);
    k_trig1<<<cdiv((long long)BATCH * 784 * C1_T, TB), TB, 0, stream>>>(x, feat1);
    k_conv1_pool<<<cdiv((long long)BATCH * NPIX * C1_OC, TB), TB, 0, stream>>>(
        feat1, w1f, c1b, h1);

    // stage 2: conv2 (WMMA)
    k_fold_w2<<<cdiv((long long)C2_M * C2_POS * C2_TP, TB), TB, 0, stream>>>(c2c, w2f);
    k_trig2<<<cdiv((long long)BATCH * NPIX * C2_TP, TB), TB, 0, stream>>>(h1, feat2);
    k_conv2_wmma<<<NPIX_ALL / 256, TB, 0, stream>>>(feat2, w2f, out2);
    k_pool2_flat<<<cdiv((long long)BATCH * C2_OC * 49, TB), TB, 0, stream>>>(
        out2, c2b, h2);

    // stage 3: fk1 (WMMA)
    k_trig3<<<cdiv((long long)BATCH * FK1_KP, TB), TB, 0, stream>>>(h2, feat3);
    k_fold_w3<<<cdiv((long long)FK1_OP * FK1_KP, TB), TB, 0, stream>>>(fk1c, wf1);
    k_fk1_wmma<<<12, TB, 0, stream>>>(feat3, wf1, fk1b, y1);

    // stage 4: fk2
    k_fk2<<<cdiv(BATCH * FK2_O, TB), TB, 0, stream>>>(y1, fk2c, fk2b, out);
}
